// Scaled_Dot_Product_Attention_19499151523893
// MI455X (gfx1250) — matrix operand order repaired and checked by reading, not compiled, not run on hardware
//
#include <hip/hip_runtime.h>

typedef __attribute__((ext_vector_type(16))) _Float16 v16h;
typedef __attribute__((ext_vector_type(8)))  float    v8f;

typedef int v4i __attribute__((vector_size(16)));
typedef __attribute__((address_space(1))) v4i* as1_v4i_ptr;  // global int4*
typedef __attribute__((address_space(3))) v4i* as3_v4i_ptr;  // LDS int4*

#if defined(__has_builtin)
#if __has_builtin(__builtin_amdgcn_global_load_async_to_lds_b128)
#define HAVE_ASYNC_COPY 1
#endif
#endif

#ifndef HAVE_ASYNC_COPY
#define HAVE_ASYNC_COPY 0
#endif

#define SEQ   1024
#define DH    64
#define QT    16          // q rows per workgroup
#define WAVES 4
#define NTHREADS 128
#define TILES_PER_WAVE 16 // 64 key tiles / 4 waves
#define KCHUNKS 32        // SEQ / 32

__device__ __forceinline__ void wait_async_lds() {
#if HAVE_ASYNC_COPY
#if __has_builtin(__builtin_amdgcn_s_wait_asynccnt)
    __builtin_amdgcn_s_wait_asynccnt(0);
#else
    asm volatile("s_wait_asynccnt 0" ::: "memory");
#endif
#endif
}

__global__ __launch_bounds__(NTHREADS)
void sdpa_fused_wmma(const float* __restrict__ Q, const float* __restrict__ K,
                     const float* __restrict__ V, float* __restrict__ Out,
                     float* __restrict__ Pattn)
{
    __shared__ _Float16 Qlds[QT][DH];       // 2 KB
    __shared__ _Float16 Plds[QT][SEQ];      // 32 KB
    __shared__ float    Vlds[32 * DH];      // 8 KB (fp32; async copy can't convert)
    __shared__ float    red[WAVES][16];
    __shared__ float    rowred[16];

    const int tid  = threadIdx.x;
    const int wave = tid >> 5;
    const int lane = tid & 31;
    const int ln   = lane & 15;   // column / row-in-half index
    const int hi   = lane >> 4;   // which 16-lane half

    const int bh    = blockIdx.x >> 6;  // b*H + h  (0..63)
    const int qblk  = blockIdx.x & 63;  // q tile   (0..63)
    const int qbase = qblk * QT;

    const float* Qp = Q + ((size_t)bh * SEQ + qbase) * DH;
    const float* Kp = K + (size_t)bh * SEQ * DH;
    const float* Vp = V + (size_t)bh * SEQ * DH;
    float* Op = Out   + ((size_t)bh * SEQ + qbase) * DH;
    float* Pp = Pattn + ((size_t)bh * SEQ + qbase) * SEQ;

    // ---- stage Q tile into LDS as f16 (coalesced) ----
    #pragma unroll
    for (int i = 0; i < (QT * DH) / NTHREADS; ++i) {
        int idx = tid + i * NTHREADS;
        Qlds[idx / DH][idx % DH] = (_Float16)Qp[idx];
    }
    __syncthreads();

    // ---- build Q A-fragments (16x32 f16 layout: lane=row M, halves hold
    //      K {0..15} / {16..31}); two K=32 chunks cover D=64 ----
    v16h aq0, aq1;
    #pragma unroll
    for (int e = 0; e < 8; ++e) {
        aq0[e]     = Qlds[ln][hi * 16 + e];
        aq0[e + 8] = Qlds[ln][hi * 16 + 8 + e];
        aq1[e]     = Qlds[ln][32 + hi * 16 + e];
        aq1[e + 8] = Qlds[ln][32 + hi * 16 + 8 + e];
    }

    // ---- phase 1: scores = Q K^T (16 tiles of 16x16 per wave) ----
    v8f sc[TILES_PER_WAVE];
    #pragma unroll
    for (int j = 0; j < TILES_PER_WAVE; ++j) {
        const int n0 = (j * WAVES + wave) * 16;
        // B-fragment: B[k][n] = K[n0+n][k] -> lane ln reads 16 contiguous floats
        const float* kr = Kp + (size_t)(n0 + ln) * DH + hi * 16;
        if (j + 1 < TILES_PER_WAVE) {
            // prefetch next tile's K row into cache (global_prefetch_b8)
            __builtin_prefetch(Kp + (size_t)(((j + 1) * WAVES + wave) * 16 + ln) * DH, 0, 3);
        }
        v16h b0, b1;
        #pragma unroll
        for (int e = 0; e < 16; ++e) {
            b0[e] = (_Float16)kr[e];        // d in [hi*16, hi*16+16)
            b1[e] = (_Float16)kr[32 + e];   // d in [32+hi*16, ...)
        }
        v8f c = {};
        c = __builtin_amdgcn_wmma_f32_16x16x32_f16(false, aq0, false, b0, (short)0, c, false, false);
        c = __builtin_amdgcn_wmma_f32_16x16x32_f16(false, aq1, false, b1, (short)0, c, false, false);
        sc[j] = c;
    }

    // ---- row max: elementwise across tiles, butterfly across 16-lane halves ----
    v8f m = sc[0];
    #pragma unroll
    for (int j = 1; j < TILES_PER_WAVE; ++j)
        #pragma unroll
        for (int e = 0; e < 8; ++e) m[e] = fmaxf(m[e], sc[j][e]);
    #pragma unroll
    for (int mask = 1; mask <= 8; mask <<= 1)
        #pragma unroll
        for (int e = 0; e < 8; ++e) m[e] = fmaxf(m[e], __shfl_xor(m[e], mask, 32));
    if (lane == 0 || lane == 16) {
        #pragma unroll
        for (int e = 0; e < 8; ++e) red[wave][hi * 8 + e] = m[e];
    }
    __syncthreads();
    if (tid < 16) {
        float v = red[0][tid];
        for (int w = 1; w < WAVES; ++w) v = fmaxf(v, red[w][tid]);
        rowred[tid] = v;
    }
    __syncthreads();
    v8f rmax;
    #pragma unroll
    for (int e = 0; e < 8; ++e) rmax[e] = rowred[hi * 8 + e];

    // ---- exp((s - max) * 1/sqrt(D)) and row sums ----
    v8f ssum = {};
    #pragma unroll
    for (int j = 0; j < TILES_PER_WAVE; ++j)
        #pragma unroll
        for (int e = 0; e < 8; ++e) {
            float p = __expf((sc[j][e] - rmax[e]) * 0.125f);
            sc[j][e] = p;
            ssum[e] += p;
        }
    #pragma unroll
    for (int mask = 1; mask <= 8; mask <<= 1)
        #pragma unroll
        for (int e = 0; e < 8; ++e) ssum[e] += __shfl_xor(ssum[e], mask, 32);
    __syncthreads();
    if (lane == 0 || lane == 16) {
        #pragma unroll
        for (int e = 0; e < 8; ++e) red[wave][hi * 8 + e] = ssum[e];
    }
    __syncthreads();
    if (tid < 16) {
        float v = red[0][tid];
        for (int w = 1; w < WAVES; ++w) v += red[w][tid];
        rowred[tid] = v;
    }
    __syncthreads();
    v8f rinv;
    #pragma unroll
    for (int e = 0; e < 8; ++e) rinv[e] = 1.0f / rowred[hi * 8 + e];

    // ---- normalize, write p_attn (fp32, non-temporal) + P f16 to LDS ----
    #pragma unroll
    for (int j = 0; j < TILES_PER_WAVE; ++j) {
        const int n0 = (j * WAVES + wave) * 16;
        #pragma unroll
        for (int e = 0; e < 8; ++e) {
            float p = sc[j][e] * rinv[e];
            int   qr = e + 8 * hi;           // C layout: M = e + 8*hi, N = ln
            __builtin_nontemporal_store(p, &Pp[(size_t)qr * SEQ + n0 + ln]);
            Plds[qr][n0 + ln] = (_Float16)p;
        }
    }
    __syncthreads();

    // ---- phase 2: out = P V ; wave owns d-tile [d0, d0+16) ----
    const int d0 = wave * 16;
    v8f oacc = {};
    for (int ch = 0; ch < KCHUNKS; ++ch) {
        const int k0 = ch * 32;
        const float* gsrc = Vp + (size_t)k0 * DH;
#if HAVE_ASYNC_COPY
        // async global->LDS: 8 KB chunk, 16 B per lane per issue, no VGPR data
        #pragma unroll
        for (int i = 0; i < (32 * DH) / (NTHREADS * 4); ++i) {
            int idx = (tid + i * NTHREADS) * 4;   // float index, 16B granules
            __builtin_amdgcn_global_load_async_to_lds_b128(
                (as1_v4i_ptr)(gsrc + idx),
                (as3_v4i_ptr)(&Vlds[idx]),
                0, 0);
        }
        wait_async_lds();
#else
        #pragma unroll
        for (int i = 0; i < (32 * DH) / (NTHREADS * 4); ++i) {
            int idx = (tid + i * NTHREADS) * 4;
            *(float4*)&Vlds[idx] = *(const float4*)(gsrc + idx);
        }
#endif
        __syncthreads();
        v16h ap, bv;
        #pragma unroll
        for (int e = 0; e < 8; ++e) {
            ap[e]     = Plds[ln][k0 + hi * 16 + e];
            ap[e + 8] = Plds[ln][k0 + hi * 16 + 8 + e];
        }
        #pragma unroll
        for (int e = 0; e < 16; ++e)
            bv[e] = (_Float16)Vlds[(hi * 16 + e) * DH + d0 + ln]; // B[k][n]=V[k0+k][d0+n]
        oacc = __builtin_amdgcn_wmma_f32_16x16x32_f16(false, ap, false, bv, (short)0, oacc, false, false);
        __syncthreads();   // all reads done before next chunk's async writes land
    }
    #pragma unroll
    for (int e = 0; e < 8; ++e)
        __builtin_nontemporal_store(oacc[e], &Op[(size_t)(e + 8 * hi) * DH + d0 + ln]);
}

extern "C" void kernel_launch(void* const* d_in, const int* in_sizes, int n_in,
                              void* d_out, int out_size, void* d_ws, size_t ws_size,
                              hipStream_t stream) {
    (void)in_sizes; (void)n_in; (void)d_ws; (void)ws_size; (void)out_size;
    const float* Q = (const float*)d_in[0];
    const float* K = (const float*)d_in[1];
    const float* V = (const float*)d_in[2];
    // d_in[3] = attn_mask (unused by reference), d_in[4] = d_k (=64, hardcoded)

    float* Out   = (float*)d_out;                              // [4,16,1024,64]
    float* Pattn = (float*)d_out + (size_t)4 * 16 * 1024 * 64; // [4,16,1024,1024]

    const int nblocks = 4 * 16 * (SEQ / QT);  // 64 (b,h) * 64 q-tiles = 4096
    sdpa_fused_wmma<<<nblocks, NTHREADS, 0, stream>>>(Q, K, V, Out, Pattn);
}
